// MMT_77481210019969
// MI455X (gfx1250) — matrix operand order repaired and checked by reading, not compiled, not run on hardware
//
#include <hip/hip_runtime.h>

typedef __attribute__((ext_vector_type(16))) _Float16 v16h;
typedef __attribute__((ext_vector_type(8)))  float    v8f;
typedef __attribute__((ext_vector_type(2)))  __fp16   pkh2;   // cvt_pkrtz result type

#define B_TOT   1024
#define NCON    3
#define NWIN    49
#define NTOK    147          // NCON * NWIN
#define NPAD    160          // NTOK padded to 16
#define DIMC    192
#define NHEAD   6
#define HDIM    32
#define NMASK   64
#define QK_SCALE 0.17677669529663687f   // 32^-0.5

// ---- workspace layout (units: _Float16 elements) ----
#define WT_OFF  ((size_t)0)
#define WT_SZ   ((size_t)4 * DIMC * DIMC)                    // 4 transposed f16 weights
#define Q_OFF   (WT_OFF + WT_SZ)
#define QKV_SZ  ((size_t)B_TOT * NHEAD * NTOK * HDIM)        // [b][h][n][d]
#define K_OFF   (Q_OFF + QKV_SZ)
#define V_OFF   (K_OFF + QKV_SZ)
#define BM_OFF  (V_OFF + QKV_SZ)
#define BM_SZ   ((size_t)NMASK * NHEAD * NTOK * NTOK)        // bias+mask table
#define AO_OFF  (BM_OFF + BM_SZ)                             // attn output f16 (B*147,192)

// ------------------------------------------------------------------
// CDNA5 helpers
// ------------------------------------------------------------------
__device__ __forceinline__ v8f wmma_f16(v16h a, v16h b, v8f c) {
  return __builtin_amdgcn_wmma_f32_16x16x32_f16(false, a, false, b, (short)0, c,
                                                false, false);
}

// Async global->LDS 16B copy (GV mode, per-lane). Tracked by ASYNCcnt.
__device__ __forceinline__ void async_b128(const void* lds_dst, const void* gsrc) {
  unsigned lds_off = (unsigned)(unsigned long long)lds_dst;   // low 32b = LDS offset
  asm volatile("global_load_async_to_lds_b128 %0, %1, off"
               :: "v"(lds_off), "v"(gsrc) : "memory");
}
__device__ __forceinline__ void wait_async0() {
  asm volatile("s_wait_asynccnt 0x0" ::: "memory");
}

// A matrix 16x32 f16: lane L -> row M=L&15; half-slot interleave per ISA 7.12.2
__device__ __forceinline__ v16h load_a_frag(const _Float16* base, int stride_h, int lane) {
  const int m   = lane & 15;
  const int hi8 = (lane >> 4) << 3;
  const unsigned int* p = (const unsigned int*)(base + (size_t)m * stride_h);
  union { v16h v; unsigned int u[8]; } r;
#pragma unroll
  for (int t = 0; t < 8; ++t) {
    const int k0 = 2 * (t & 3) + hi8 + ((t >> 2) << 4);     // even; pairs (k0,k0+1)
    r.u[t] = p[k0 >> 1];
  }
  return r.v;
}

// B matrix 32x16 f16: lane L -> column N=L&15; 16 half-slots = K in same interleave as A
__device__ __forceinline__ v16h load_b_frag(const _Float16* base, int stride_h, int lane) {
  const int n   = lane & 15;
  const int hi8 = (lane >> 4) << 3;
  union { v16h v; _Float16 e[16]; } r;
#pragma unroll
  for (int s = 0; s < 16; ++s) {
    const int k = (s & 7) + hi8 + ((s >> 3) << 4);
    r.e[s] = base[(size_t)k * stride_h + n];
  }
  return r.v;
}

// ------------------------------------------------------------------
// Prep 1: convert 4 weight matrices (out,in)->f16 transposed (k-major)
// ------------------------------------------------------------------
__global__ void prep_w_kernel(const float* qw, const float* kw, const float* vw,
                              const float* pw, _Float16* ws) {
  int idx = blockIdx.x * blockDim.x + threadIdx.x;
  if (idx >= 4 * DIMC * DIMC) return;
  int mat = idx / (DIMC * DIMC);
  int r   = idx % (DIMC * DIMC);
  int k = r / DIMC, n = r % DIMC;
  const float* src = (mat == 0) ? qw : (mat == 1) ? kw : (mat == 2) ? vw : pw;
  ws[WT_OFF + (size_t)mat * DIMC * DIMC + (size_t)k * DIMC + n] =
      (_Float16)src[(size_t)n * DIMC + k];
}

// ------------------------------------------------------------------
// Prep 2: combined relative-position bias + window mask table
// bm[w][h][i][j] = mask[w][i%49][j%49] + bias_table[rel(i%49,j%49)*6 + h]
// ------------------------------------------------------------------
__global__ void prep_bm_kernel(const float* mask, const float* bias_table, _Float16* ws) {
  long long idx = (long long)blockIdx.x * blockDim.x + threadIdx.x;
  if (idx >= (long long)NMASK * NHEAD * NTOK * NTOK) return;
  int j = (int)(idx % NTOK);
  long long t = idx / NTOK;
  int i = (int)(t % NTOK);
  t /= NTOK;
  int h = (int)(t % NHEAD);
  int w = (int)(t / NHEAD);
  int iw = i % NWIN, jw = j % NWIN;
  int ri = iw / 7, ci = iw % 7, rj = jw / 7, cj = jw % 7;
  int rel = (ri - rj + 6) * 13 + (ci - cj + 6);
  float v = mask[((size_t)w * NWIN + iw) * NWIN + jw] + bias_table[rel * NHEAD + h];
  ws[BM_OFF + ((size_t)(w * NHEAD + h) * NTOK + i) * NTOK + j] = (_Float16)v;
}

// ------------------------------------------------------------------
// Kernel 1: fused QKV projection.  out = x @ W^T + b  (q also * scale)
// grid=(1176,3) block=256 (8 waves). 128-row macro tile, wave owns 16 rows x 192 cols.
// ------------------------------------------------------------------
__global__ void __launch_bounds__(256)
qkv_proj_kernel(const float* xq, const float* xk, const float* xv,
                const float* qb, const float* kb, const float* vb,
                _Float16* ws) {
  __shared__ __align__(16) _Float16 Xs[128 * 40];   // 128 rows x 32 k (+8 pad)
  __shared__ __align__(16) _Float16 Wk[32 * 200];   // 32 k-rows x 192 n (+8 pad)

  const int s    = blockIdx.y;                      // 0=q 1=k 2=v
  const int M0   = blockIdx.x * 128;
  const int tid  = threadIdx.x;
  const int wave = tid >> 5, lane = tid & 31;

  const float*    x  = (s == 0) ? xq : (s == 1) ? xk : xv;
  const float*    bi = (s == 0) ? qb : (s == 1) ? kb : vb;
  const _Float16* wt = ws + WT_OFF + (size_t)s * DIMC * DIMC;
  _Float16*       ob = ws + Q_OFF + (size_t)s * QKV_SZ;
  const float scale  = (s == 0) ? QK_SCALE : 1.0f;

  v8f acc[12];
#pragma unroll
  for (int i = 0; i < 12; ++i) acc[i] = 0;

  for (int kt = 0; kt < 6; ++kt) {
    const int k0 = kt * 32;
    // stage W^T slab via async global->LDS (pure f16 copy), 768 x 16B
    for (int idx = tid; idx < 768; idx += 256) {
      int row = idx / 24, c8 = (idx % 24) << 3;
      async_b128(Wk + row * 200 + c8, wt + (size_t)(k0 + row) * DIMC + c8);
    }
    // stage X tile (f32 -> f16, packed cvt + 8B DS stores), 1024 float4 loads
    for (int idx = tid; idx < 1024; idx += 256) {
      int row = idx >> 3, c4 = (idx & 7) << 2;
      const float4 v = *(const float4*)(x + (size_t)(M0 + row) * DIMC + k0 + c4);
      union { pkh2 h[2]; uint2 u; } pk;
      pk.h[0] = __builtin_amdgcn_cvt_pkrtz(v.x, v.y);
      pk.h[1] = __builtin_amdgcn_cvt_pkrtz(v.z, v.w);
      *(uint2*)(Xs + row * 40 + c4) = pk.u;
    }
    wait_async0();
    __syncthreads();

    v16h a = load_a_frag(Xs + wave * 16 * 40, 40, lane);
#pragma unroll
    for (int nt = 0; nt < 12; ++nt) {
      v16h b = load_b_frag(Wk + nt * 16, 200, lane);
      acc[nt] = wmma_f16(a, b, acc[nt]);
    }
    __syncthreads();
  }

  // epilogue: add bias, (scale), scatter to [b][h][n][d] f16
  const int li = lane & 15, hi = lane >> 4;
#pragma unroll
  for (int r = 0; r < 8; ++r) {
    const int m = M0 + wave * 16 + r + 8 * hi;
    const int bb = m / NTOK, n = m % NTOK;
#pragma unroll
    for (int nt = 0; nt < 12; ++nt) {
      const int c = nt * 16 + li;
      const int h = c >> 5, d = c & 31;
      float v = (acc[nt][r] + bi[c]) * scale;
      ob[((size_t)(bb * NHEAD + h) * NTOK + n) * HDIM + d] = (_Float16)v;
    }
  }
}

// ------------------------------------------------------------------
// Kernel 2: attention per (b,h).  block=320 (10 waves), wave = one 16-row i-tile.
// logits = Q@K^T (+bm), softmax (fragment-layout reductions), P@V.
// ------------------------------------------------------------------
__global__ void __launch_bounds__(320)
attn_kernel(_Float16* ws) {
  extern __shared__ __align__(16) _Float16 smem[];
  _Float16* Qs = smem;              // [160][40]
  _Float16* Kt = smem + 6400;       // [32][168]  (Kt[d][j])
  _Float16* Vs = smem + 11776;      // [160][40]
  _Float16* As = smem + 18176;      // [160][168] softmaxed probs

  const int b = blockIdx.x / NHEAD, h = blockIdx.x % NHEAD;
  const int tid = threadIdx.x, wave = tid >> 5, lane = tid & 31;
  const _Float16* qp = ws + Q_OFF + (size_t)(b * NHEAD + h) * NTOK * HDIM;
  const _Float16* kp = ws + K_OFF + (size_t)(b * NHEAD + h) * NTOK * HDIM;
  const _Float16* vp = ws + V_OFF + (size_t)(b * NHEAD + h) * NTOK * HDIM;

  // stage Q and V via async global->LDS (147*32 halfs = 588 uint4 each)
  for (int idx = tid; idx < 588; idx += 320) {
    int row = idx >> 2, c8 = (idx & 3) << 3;
    async_b128(Qs + row * 40 + c8, qp + row * HDIM + c8);
    async_b128(Vs + row * 40 + c8, vp + row * HDIM + c8);
  }
  // zero pad rows 147..159
  for (int idx = tid; idx < 13 * 40; idx += 320) {
    Qs[(NTOK + idx / 40) * 40 + idx % 40] = (_Float16)0.0f;
    Vs[(NTOK + idx / 40) * 40 + idx % 40] = (_Float16)0.0f;
  }
  // stage K transposed: Kt[d][j]
  for (int idx = tid; idx < 32 * NPAD; idx += 320) {
    int j = idx >> 5, d = idx & 31;
    Kt[d * 168 + j] = (j < NTOK) ? kp[j * HDIM + d] : (_Float16)0.0f;
  }
  wait_async0();
  __syncthreads();

  // ---- logits: 16x160 per wave ----
  const int i0 = wave * 16;
  v8f s_acc[10];
#pragma unroll
  for (int jt = 0; jt < 10; ++jt) s_acc[jt] = 0;

  v16h aq = load_a_frag(Qs + i0 * 40, 40, lane);
#pragma unroll
  for (int jt = 0; jt < 10; ++jt) {
    v16h bk = load_b_frag(Kt + jt * 16, 168, lane);
    s_acc[jt] = wmma_f16(aq, bk, s_acc[jt]);
  }

  // ---- bias+mask + softmax on fragments ----
  const int li = lane & 15, hi = lane >> 4;
  const _Float16* bmp = ws + BM_OFF + (size_t)((b & (NMASK - 1)) * NHEAD + h) * NTOK * NTOK;
#pragma unroll
  for (int r = 0; r < 8; ++r) {
    const int i  = i0 + r + 8 * hi;
    const int ic = (i < NTOK) ? i : (NTOK - 1);
    float mx = -3.0e38f;
#pragma unroll
    for (int jt = 0; jt < 10; ++jt) {
      const int j = jt * 16 + li;
      float addv = (j < NTOK) ? (float)bmp[(size_t)ic * NTOK + j] : -3.0e38f;
      float v = s_acc[jt][r] + addv;
      s_acc[jt][r] = v;
      mx = fmaxf(mx, v);
    }
#pragma unroll
    for (int o = 1; o < 16; o <<= 1) mx = fmaxf(mx, __shfl_xor(mx, o, 16));
    float sum = 0.0f;
#pragma unroll
    for (int jt = 0; jt < 10; ++jt) {
      float e = __expf(s_acc[jt][r] - mx);
      s_acc[jt][r] = e;
      sum += e;
    }
#pragma unroll
    for (int o = 1; o < 16; o <<= 1) sum += __shfl_xor(sum, o, 16);
    const float inv = 1.0f / sum;
#pragma unroll
    for (int jt = 0; jt < 10; ++jt)
      As[(size_t)i * 168 + jt * 16 + li] = (_Float16)(s_acc[jt][r] * inv);
  }
  // LDS ops within a wave are in-order; drain before reading probs back.
  asm volatile("s_wait_dscnt 0x0" ::: "memory");

  // ---- out = P @ V : 16x32 per wave ----
  v8f o_acc[2];
  o_acc[0] = 0; o_acc[1] = 0;
#pragma unroll
  for (int ks = 0; ks < 5; ++ks) {
    v16h ap = load_a_frag(As + (size_t)i0 * 168 + ks * 32, 168, lane);
#pragma unroll
    for (int dt = 0; dt < 2; ++dt) {
      v16h bv = load_b_frag(Vs + ks * 32 * 40 + dt * 16, 40, lane);
      o_acc[dt] = wmma_f16(ap, bv, o_acc[dt]);
    }
  }
  _Float16* ao = ws + AO_OFF;
#pragma unroll
  for (int dt = 0; dt < 2; ++dt) {
    const int d = dt * 16 + li;
#pragma unroll
    for (int r = 0; r < 8; ++r) {
      const int i = i0 + r + 8 * hi;
      if (i < NTOK)
        ao[((size_t)b * NTOK + i) * DIMC + h * HDIM + d] = (_Float16)o_acc[dt][r];
    }
  }
}

// ------------------------------------------------------------------
// Kernel 3: output projection.  d_out = ao @ proj_w^T + proj_b (f32 out)
// ------------------------------------------------------------------
__global__ void __launch_bounds__(256)
proj_kernel(const _Float16* ws, const float* pb, float* out) {
  __shared__ __align__(16) _Float16 Xs[128 * 40];
  __shared__ __align__(16) _Float16 Wk[32 * 200];

  const int M0 = blockIdx.x * 128;
  const int tid = threadIdx.x, wave = tid >> 5, lane = tid & 31;
  const _Float16* ao = ws + AO_OFF;
  const _Float16* wt = ws + WT_OFF + (size_t)3 * DIMC * DIMC;

  v8f acc[12];
#pragma unroll
  for (int i = 0; i < 12; ++i) acc[i] = 0;

  for (int kt = 0; kt < 6; ++kt) {
    const int k0 = kt * 32;
    // both stages are pure f16 copies -> async global->LDS
    for (int idx = tid; idx < 512; idx += 256) {   // 128x32 halfs = 512 uint4
      int row = idx >> 2, c8 = (idx & 3) << 3;
      async_b128(Xs + row * 40 + c8, ao + (size_t)(M0 + row) * DIMC + k0 + c8);
    }
    for (int idx = tid; idx < 768; idx += 256) {
      int row = idx / 24, c8 = (idx % 24) << 3;
      async_b128(Wk + row * 200 + c8, wt + (size_t)(k0 + row) * DIMC + c8);
    }
    wait_async0();
    __syncthreads();

    v16h a = load_a_frag(Xs + wave * 16 * 40, 40, lane);
#pragma unroll
    for (int nt = 0; nt < 12; ++nt) {
      v16h b = load_b_frag(Wk + nt * 16, 200, lane);
      acc[nt] = wmma_f16(a, b, acc[nt]);
    }
    __syncthreads();
  }

  const int li = lane & 15, hi = lane >> 4;
#pragma unroll
  for (int r = 0; r < 8; ++r) {
    const int m = M0 + wave * 16 + r + 8 * hi;
#pragma unroll
    for (int nt = 0; nt < 12; ++nt) {
      const int c = nt * 16 + li;
      out[(size_t)m * DIMC + c] = acc[nt][r] + pb[c];
    }
  }
}

// ------------------------------------------------------------------
extern "C" void kernel_launch(void* const* d_in, const int* in_sizes, int n_in,
                              void* d_out, int out_size, void* d_ws, size_t ws_size,
                              hipStream_t stream) {
  const float* xq = (const float*)d_in[0];
  const float* xk = (const float*)d_in[1];
  const float* xv = (const float*)d_in[2];
  const float* mask = (const float*)d_in[3];
  const float* qw = (const float*)d_in[4];
  const float* qb = (const float*)d_in[5];
  const float* kw = (const float*)d_in[6];
  const float* kb = (const float*)d_in[7];
  const float* vw = (const float*)d_in[8];
  const float* vb = (const float*)d_in[9];
  const float* pw = (const float*)d_in[10];
  const float* pb = (const float*)d_in[11];
  const float* bt = (const float*)d_in[12];
  _Float16* ws = (_Float16*)d_ws;
  float* out = (float*)d_out;

  prep_w_kernel<<<(4 * DIMC * DIMC + 255) / 256, 256, 0, stream>>>(qw, kw, vw, pw, ws);

  long long bm_total = (long long)NMASK * NHEAD * NTOK * NTOK;
  prep_bm_kernel<<<(unsigned)((bm_total + 255) / 256), 256, 0, stream>>>(mask, bt, ws);

  dim3 g1((B_TOT * NTOK) / 128, 3);
  qkv_proj_kernel<<<g1, 256, 0, stream>>>(xq, xk, xv, qb, kb, vb, ws);

  const size_t attn_lds = (size_t)(6400 + 5376 + 6400 + 26880) * sizeof(_Float16); // 90112 B
  attn_kernel<<<B_TOT * NHEAD, 320, attn_lds, stream>>>(ws);

  proj_kernel<<<(B_TOT * NTOK) / 128, 256, 0, stream>>>(ws, pb, out);
}
